// CNN_RNN_75823352644233
// MI455X (gfx1250) — matrix operand order repaired and checked by reading, not compiled, not run on hardware
//
#include <hip/hip_runtime.h>
#include <hip/hip_bf16.h>

// ---------------------------------------------------------------------------
// CDNA5 (gfx1250) implementation of the CNN+BiGRU sentence extractor.
// All GEMMs (conv1/conv2 via layout-free im2col, GRU input proj, GRU
// recurrence) run on v_wmma_f32_16x16x32_f16.  f16 operands, f32 accum.
// Conv activation panels are staged into LDS with the gfx1250 async
// global->LDS engine (ASYNCcnt) when the toolchain exposes it.
// ---------------------------------------------------------------------------

typedef __attribute__((ext_vector_type(16))) _Float16 v16h;
typedef __attribute__((ext_vector_type(8)))  float    v8f;
typedef __attribute__((ext_vector_type(4)))  int      vi4;

// ---- WMMA fragment helpers (wave32, this file's contiguous k = 16*(lane/16)+e order) --------------------
// A 16x32 (MxK), row-major source, row stride lda:
//   lane<16 : M=lane,    elems 0..15 -> K=0..15
//   lane>=16: M=lane-16, elems 0..15 -> K=16..31
__device__ __forceinline__ v16h frag_a_load(const _Float16* __restrict__ p, int lda) {
  const int lane = threadIdx.x & 31;
  const int m    = lane & 15;
  const int kb   = (lane >> 4) << 4;            // 0 or 16
  const _Float16* r = p + m * lda + kb;
  v16h a;
#pragma unroll
  for (int e = 0; e < 16; ++e) a[e] = r[e];
  return a;
}

// B 32x16 (KxN) supplied as B^T row-major (row n contiguous over K), stride ldb:
//   lane<16 : N=lane,    elems 0..15 -> K=0..15
//   lane>=16: N=lane-16, elems 0..15 -> K=16..31
__device__ __forceinline__ v16h frag_b_load(const _Float16* __restrict__ p, int ldb) {
  const int lane = threadIdx.x & 31;
  const int n    = lane & 15;
  const int kb   = (lane >> 4) << 4;            // 0 or 16
  const _Float16* r = p + n * ldb + kb;
  v16h b;
#pragma unroll
  for (int e = 0; e < 16; ++e) b[e] = r[e];
  return b;
}

// C/D 16x16 f32: vgpr r holds M = r + 8*(lane>=16), N = lane&15
__device__ __forceinline__ void frag_c_store(float* __restrict__ p, int ldc, v8f c) {
  const int lane = threadIdx.x & 31;
  const int n    = lane & 15;
  const int mb   = (lane >> 4) << 3;
#pragma unroll
  for (int r = 0; r < 8; ++r) p[(mb + r) * ldc + n] = c[r];
}

__device__ __forceinline__ v8f wmma16(v16h a, v16h b, v8f c) {
  return __builtin_amdgcn_wmma_f32_16x16x32_f16(false, a, false, b, (short)0, c,
                                                false, false);
}

// ---- async global->LDS staging helpers ------------------------------------
__device__ __forceinline__ void g2l_b128(_Float16* l, const _Float16* g) {
#if defined(__HIP_DEVICE_COMPILE__) && __has_builtin(__builtin_amdgcn_global_load_async_to_lds_b128)
  // GLOBAL_LOAD_ASYNC_TO_LDS_B128: per-lane 16B memory -> LDS, ASYNCcnt-tracked.
  // Param 1 is AS(1) ("__device__") non-const ptr to int ext_vector(4), per
  // the toolchain's own diagnostic; param 2 symmetric in AS(3).
  __builtin_amdgcn_global_load_async_to_lds_b128(
      (__attribute__((address_space(1))) vi4*)g,
      (__attribute__((address_space(3))) vi4*)l, 0, 0);
#else
  *(uint4*)l = *(const uint4*)g;                // fallback: load + ds_store
#endif
}

__device__ __forceinline__ void async_wait0() {
#if defined(__HIP_DEVICE_COMPILE__)
#if __has_builtin(__builtin_amdgcn_s_wait_asynccnt)
  __builtin_amdgcn_s_wait_asynccnt(0);
#else
  asm volatile("s_wait_asynccnt 0x0" ::: "memory");
#endif
#endif
}

// ---------------------------------------------------------------------------
// Conv layer GEMM with LDS-staged activation panel.
// One block per sentence.  The K*D-contiguous patch layout makes im2col a
// sliding window: panel[t*128 .. t*128+Kd) is the conv patch at position t.
// C'[t][co] = panel_patch(t) . wflat(co);  M=48, N=112, lda=128 (overlapped).
// ---------------------------------------------------------------------------
#define PANEL_H 6656                             // 48*128 + 512 slack (halfs)
__global__ __launch_bounds__(256) void conv_gemm_lds(
    const _Float16* __restrict__ A, long astride,
    const _Float16* __restrict__ Bt, int Kd,
    float* __restrict__ C, long cstride)
{
  __shared__ __align__(16) _Float16 panel[PANEL_H];     // 13 KB
  const _Float16* Ab = A + (long)blockIdx.x * astride;

  // stage 6656 halfs = 832 x b128 transfers, async -> LDS
  for (int i = threadIdx.x; i < PANEL_H / 8; i += 256)
    g2l_b128(&panel[i * 8], Ab + i * 8);
  async_wait0();
  __syncthreads();

  const int wid = threadIdx.x >> 5;              // 8 waves, 21 tiles (3m x 7n)
  float* Cb = C + (long)blockIdx.x * cstride;
  for (int tl = wid; tl < 21; tl += 8) {
    const int mt = tl % 3;
    const int nt = tl / 3;
    const _Float16* ap = &panel[(mt << 4) * 128];
    const _Float16* bp = Bt + (long)(nt << 4) * Kd;
    v8f acc = {};
    for (int k = 0; k < Kd; k += 32) {
      v16h af = frag_a_load(ap + k, 128);        // ds_load fragments
      v16h bf = frag_b_load(bp + k, Kd);         // weights: L2-hot global
      acc = wmma16(af, bf, acc);
    }
    frag_c_store(Cb + (long)(mt << 4) * 112 + (nt << 4), 112, acc);
  }
}

// ---------------------------------------------------------------------------
// Generic GEMM: C (MxN f32, ldc) = A (MxKd f16, row stride lda)
//                                x Bt (NxKd f16, row stride ldb)^T
// One wave per 16x16 tile, grid-stride.  Used for the GRU input projections.
// ---------------------------------------------------------------------------
__global__ __launch_bounds__(256) void gemm_bt(
    const _Float16* __restrict__ A, int lda,
    const _Float16* __restrict__ Bt, int ldb,
    float* __restrict__ C, int ldc,
    int M, int N, int Kd)
{
  const int mtiles = M >> 4;
  const int tiles  = mtiles * (N >> 4);
  const int gw = (blockIdx.x * 256 + threadIdx.x) >> 5;
  const int nw = gridDim.x << 3;
  for (int tl = gw; tl < tiles; tl += nw) {
    const int mt = tl % mtiles;
    const int nt = tl / mtiles;
    const _Float16* ap = A + (long)(mt << 4) * lda;
    const _Float16* bp = Bt + (long)(nt << 4) * ldb;
    v8f acc = {};
    for (int k = 0; k < Kd; k += 32) {
      __builtin_prefetch(ap + k + 256, 0, 1);    // global_prefetch_b8
      v16h af = frag_a_load(ap + k, lda);
      v16h bf = frag_b_load(bp + k, ldb);
      acc = wmma16(af, bf, acc);
    }
    frag_c_store(C + (long)(mt << 4) * ldc + (nt << 4), ldc, acc);
  }
}

// ---------------------------------------------------------------------------
// Weight packing / casting kernels (f32 inputs -> padded f16 in workspace)
// ---------------------------------------------------------------------------
// w1: [100][128][K] -> [112][K*128], dst[co][k*128+d] = w1[co][d][k]
__global__ void pack_w1(const float* __restrict__ src, _Float16* __restrict__ dst, int K) {
  int i = blockIdx.x * blockDim.x + threadIdx.x;
  int tot = 112 * K * 128;
  if (i >= tot) return;
  int kd = i % (K * 128), co = i / (K * 128);
  int k = kd >> 7, d = kd & 127;
  float v = (co < 100) ? src[(co * 128 + d) * K + k] : 0.f;
  dst[i] = (_Float16)v;
}
// w2: [100][100][K] -> [112][K*128], dst[co][k*128+ci] = w2[co][ci][k]
__global__ void pack_w2(const float* __restrict__ src, _Float16* __restrict__ dst, int K) {
  int i = blockIdx.x * blockDim.x + threadIdx.x;
  int tot = 112 * K * 128;
  if (i >= tot) return;
  int kd = i % (K * 128), co = i / (K * 128);
  int k = kd >> 7, ci = kd & 127;
  float v = (co < 100 && ci < 100) ? src[(co * 100 + ci) * K + k] : 0.f;
  dst[i] = (_Float16)v;
}
// Wih: [768][200] -> [768][224] (zero-padded K)
__global__ void pack_wih(const float* __restrict__ src, _Float16* __restrict__ dst) {
  int i = blockIdx.x * blockDim.x + threadIdx.x;
  if (i >= 768 * 224) return;
  int c = i % 224, g = i / 224;
  dst[i] = (_Float16)((c < 200) ? src[g * 200 + c] : 0.f);
}
// Whh: [768][256] -> f16
__global__ void pack_whh(const float* __restrict__ src, _Float16* __restrict__ dst) {
  int i = blockIdx.x * blockDim.x + threadIdx.x;
  if (i < 768 * 256) dst[i] = (_Float16)src[i];
}

__global__ void zero_h16(_Float16* __restrict__ p, long n) {
  long i = blockIdx.x * (long)blockDim.x + threadIdx.x;
  if (i < n) p[i] = (_Float16)0.f;
}

// ---------------------------------------------------------------------------
// Embedding gather: x[N*L] -> ebuf[n][l*128+d] f16 (padding_idx 0 -> zeros)
// One wave per token row; lane handles 4 dims (float4 load).
// ---------------------------------------------------------------------------
__global__ void embed_gather(const int* __restrict__ x, const float* __restrict__ emb,
                             _Float16* __restrict__ ebuf) {
  long i = blockIdx.x * (long)blockDim.x + threadIdx.x;
  if (i >= 102400L * 32) return;
  int tok = (int)(i >> 5), lane = (int)(i & 31);
  int idx = x[tok];
  _Float16* dst = ebuf + (long)tok * 128 + lane * 4;
  if (idx == 0) {
    dst[0] = dst[1] = dst[2] = dst[3] = (_Float16)0.f;
  } else {
    const float4 v = *(const float4*)(emb + (long)idx * 128 + lane * 4);
    dst[0] = (_Float16)v.x; dst[1] = (_Float16)v.y;
    dst[2] = (_Float16)v.z; dst[3] = (_Float16)v.w;
  }
}

// ---------------------------------------------------------------------------
// Training-mode BatchNorm: deterministic two-pass reduction over (n, t<Lout).
// conv output buffer layout: [2048][48][112] f32 (channel minor).
// ---------------------------------------------------------------------------
#define STAT_BLOCKS 240
__global__ void bn_stats(const float* __restrict__ c, int Lout,
                         float* __restrict__ partial /* [STAT_BLOCKS][2][112] */) {
  int ch = threadIdx.x;                 // blockDim = 128, ch<112 active
  if (ch >= 112) return;
  float sum = 0.f, sq = 0.f;
  for (long row = blockIdx.x; row < 2048L * 48; row += gridDim.x) {
    int t = (int)(row % 48);
    if (t < Lout) {
      float v = c[row * 112 + ch];
      sum += v; sq += v * v;
    }
  }
  partial[(blockIdx.x * 2 + 0) * 112 + ch] = sum;
  partial[(blockIdx.x * 2 + 1) * 112 + ch] = sq;
}

__global__ void bn_finalize(const float* __restrict__ partial,
                            const float* __restrict__ g, const float* __restrict__ beta,
                            int Lout, float* __restrict__ scale, float* __restrict__ shift) {
  int ch = threadIdx.x;
  if (ch >= 112) return;
  float sum = 0.f, sq = 0.f;
  for (int b = 0; b < STAT_BLOCKS; ++b) {
    sum += partial[(2 * b + 0) * 112 + ch];
    sq  += partial[(2 * b + 1) * 112 + ch];
  }
  float cnt = 2048.f * (float)Lout;
  float m   = sum / cnt;
  float var = sq / cnt - m * m;
  float gg  = (ch < 100) ? g[ch]    : 0.f;
  float bb  = (ch < 100) ? beta[ch] : 0.f;
  float sc  = gg * rsqrtf(var + 1e-5f);
  scale[ch] = sc;
  shift[ch] = bb - m * sc;     // conv bias cancels in training-mode BN
}

// BN + leaky + cast to f16, channel-padded to 128 for the next conv GEMM.
__global__ void bn_apply(const float* __restrict__ c, const float* __restrict__ scale,
                         const float* __restrict__ shift, int Lout,
                         _Float16* __restrict__ mid /* [2048][48][128] */) {
  long i = blockIdx.x * (long)blockDim.x + threadIdx.x;
  if (i >= 2048L * 48 * 128) return;
  int  ch  = (int)(i & 127);
  long row = i >> 7;
  int  t   = (int)(row % 48);
  float v = 0.f;
  if (ch < 100 && t < Lout) {
    float y = scale[ch] * c[row * 112 + ch] + shift[ch];
    v = (y >= 0.f) ? y : 0.01f * y;
  }
  mid[i] = (_Float16)v;
}

// BN + leaky + global max-pool over time -> sentence feature column block.
__global__ void bn_pool(const float* __restrict__ c2, const float* __restrict__ scale,
                        const float* __restrict__ shift, int Lout2, int coloff,
                        _Float16* __restrict__ sent /* [2048][224] */) {
  int i = blockIdx.x * blockDim.x + threadIdx.x;
  if (i >= 2048 * 100) return;
  int co = i % 100, n = i / 100;
  const float* p = c2 + (long)n * 48 * 112 + co;
  float m = -1e30f;
  for (int t = 0; t < Lout2; ++t) {
    float y = scale[co] * p[t * 112] + shift[co];
    y = (y >= 0.f) ? y : 0.01f * y;
    m = fmaxf(m, y);
  }
  sent[n * 224 + coloff + co] = (_Float16)m;
}

// ---------------------------------------------------------------------------
// Persistent BiGRU scan.  2 blocks (dir 0 = fwd, 1 = bwd) x 512 threads.
// h [32][256] lives in LDS as f16; gh = h*Whh^T is computed with WMMA reading
// A-fragments straight from LDS (ds_load).  Wave w owns h-tile hc=w; computes
// r/z/n tiles for both 16-row batches, then applies the gate nonlinearities
// directly in C-fragment layout (torch gate order r,z,n).
// ---------------------------------------------------------------------------
__global__ __launch_bounds__(512) void gru_scan(
    const float* __restrict__ gi_f, const float* __restrict__ gi_b,
    const _Float16* __restrict__ whh_f, const _Float16* __restrict__ whh_b,
    const float* __restrict__ bih_f, const float* __restrict__ bhh_f,
    const float* __restrict__ bih_b, const float* __restrict__ bhh_b,
    float* __restrict__ hs_f, float* __restrict__ hs_b)
{
  __shared__ __align__(16) _Float16 hlds[32][256];     // 16 KB
  const int dir = blockIdx.x;
  const float*    gi  = dir ? gi_b  : gi_f;
  const _Float16* whh = dir ? whh_b : whh_f;
  const float*    bih = dir ? bih_b : bih_f;
  const float*    bhh = dir ? bhh_b : bhh_f;
  float*          hs  = dir ? hs_b  : hs_f;

  for (int i = threadIdx.x; i < 32 * 256; i += 512) (&hlds[0][0])[i] = (_Float16)0.f;
  __syncthreads();

  const int wid  = threadIdx.x >> 5;   // 0..15 == h-tile
  const int lane = threadIdx.x & 31;
  const int col  = lane & 15;
  const int mb   = (lane >> 4) << 3;
  const int hc   = wid;
  const _Float16* bpr = whh + (long)((hc << 4) + 0)   * 256;
  const _Float16* bpz = whh + (long)((hc << 4) + 256) * 256;
  const _Float16* bpn = whh + (long)((hc << 4) + 512) * 256;

  for (int step = 0; step < 64; ++step) {
    const int t = dir ? (63 - step) : step;
    v8f ar[2], az[2], an[2];
#pragma unroll
    for (int mt = 0; mt < 2; ++mt) {
      v8f r_ = {}, z_ = {}, n_ = {};
      const _Float16* ap = &hlds[mt << 4][0];
#pragma unroll
      for (int k = 0; k < 256; k += 32) {
        v16h af = frag_a_load(ap + k, 256);            // ds_load fragments
        r_ = wmma16(af, frag_b_load(bpr + k, 256), r_);
        z_ = wmma16(af, frag_b_load(bpz + k, 256), z_);
        n_ = wmma16(af, frag_b_load(bpn + k, 256), n_);
      }
      ar[mt] = r_; az[mt] = z_; an[mt] = n_;
    }
    __syncthreads();                                   // all reads of old h done
#pragma unroll
    for (int mt = 0; mt < 2; ++mt) {
#pragma unroll
      for (int rr = 0; rr < 8; ++rr) {
        const int b = (mt << 4) + mb + rr;
        const int h = (hc << 4) + col;
        const int n = (b << 6) + t;                    // sentence index b*64+t
        const float* g = gi + (long)n * 768;
        float hr = ar[mt][rr] + bhh[h];
        float hz = az[mt][rr] + bhh[256 + h];
        float hn = an[mt][rr] + bhh[512 + h];
        float r  = 1.f / (1.f + __expf(-(g[h]       + bih[h]       + hr)));
        float z  = 1.f / (1.f + __expf(-(g[256 + h] + bih[256 + h] + hz)));
        float nn = tanhf(g[512 + h] + bih[512 + h] + r * hn);
        float hnew = (1.f - z) * nn + z * (float)hlds[b][h];
        hlds[b][h] = (_Float16)hnew;                   // sole writer of (b,h)
        hs[(long)n * 256 + h] = hnew;
      }
    }
    __syncthreads();
  }
}

// logits: sigmoid( hf·cw[0:256] + hb·cw[256:512] + bias )
__global__ void logits_kernel(const float* __restrict__ hsf, const float* __restrict__ hsb,
                              const float* __restrict__ cw, const float* __restrict__ bias,
                              float* __restrict__ out) {
  int n = blockIdx.x * blockDim.x + threadIdx.x;
  if (n >= 2048) return;
  float s = bias[0];
  const float* pf = hsf + (long)n * 256;
  const float* pb = hsb + (long)n * 256;
  for (int h = 0; h < 256; ++h) s += pf[h] * cw[h];
  for (int h = 0; h < 256; ++h) s += pb[h] * cw[256 + h];
  out[n] = 1.f / (1.f + __expf(-s));
}

// ---------------------------------------------------------------------------
extern "C" void kernel_launch(void* const* d_in, const int* in_sizes, int n_in,
                              void* d_out, int out_size, void* d_ws, size_t ws_size,
                              hipStream_t stream)
{
  (void)n_in; (void)out_size; (void)ws_size;
  // ---- input wiring: detect flattening order via in_sizes ----
  const int* X = nullptr;
  const float *EMB = nullptr, *CW = nullptr, *BIAS = nullptr;
  const float *W1[2], *G1[2], *BE1[2], *W2[2], *G2[2], *BE2[2];
  const float *WIH[2], *WHH[2], *BIH[2], *BHH[2];
  if (in_sizes[0] == 102400) {
    // dict insertion order: x, embed, conv0{w1,b1,g1,be1,w2,b2,g2,be2}, conv1{...},
    // Wih_f,Whh_f,bih_f,bhh_f, Wih_b,Whh_b,bih_b,bhh_b, content_w, bias, B, T
    X = (const int*)d_in[0]; EMB = (const float*)d_in[1];
    for (int b = 0; b < 2; ++b) {
      int i = 2 + 8 * b;
      W1[b]  = (const float*)d_in[i + 0];
      G1[b]  = (const float*)d_in[i + 2];
      BE1[b] = (const float*)d_in[i + 3];
      W2[b]  = (const float*)d_in[i + 4];
      G2[b]  = (const float*)d_in[i + 6];
      BE2[b] = (const float*)d_in[i + 7];
    }
    WIH[0] = (const float*)d_in[18]; WHH[0] = (const float*)d_in[19];
    BIH[0] = (const float*)d_in[20]; BHH[0] = (const float*)d_in[21];
    WIH[1] = (const float*)d_in[22]; WHH[1] = (const float*)d_in[23];
    BIH[1] = (const float*)d_in[24]; BHH[1] = (const float*)d_in[25];
    CW = (const float*)d_in[26]; BIAS = (const float*)d_in[27];
  } else {
    // jax pytree sorted order: batch_docs, doc_len, bias, content_w,
    // conv0{b1,b2,be1,be2,g1,g2,w1,w2}, conv1{...}, embed,
    // Wih_b,Wih_f,Whh_b,Whh_f,bih_b,bih_f,bhh_b,bhh_f, x
    BIAS = (const float*)d_in[2]; CW = (const float*)d_in[3];
    for (int b = 0; b < 2; ++b) {
      int i = 4 + 8 * b;
      BE1[b] = (const float*)d_in[i + 2];
      BE2[b] = (const float*)d_in[i + 3];
      G1[b]  = (const float*)d_in[i + 4];
      G2[b]  = (const float*)d_in[i + 5];
      W1[b]  = (const float*)d_in[i + 6];
      W2[b]  = (const float*)d_in[i + 7];
    }
    EMB = (const float*)d_in[20];
    WIH[1] = (const float*)d_in[21]; WIH[0] = (const float*)d_in[22];
    WHH[1] = (const float*)d_in[23]; WHH[0] = (const float*)d_in[24];
    BIH[1] = (const float*)d_in[25]; BIH[0] = (const float*)d_in[26];
    BHH[1] = (const float*)d_in[27]; BHH[0] = (const float*)d_in[28];
    X = (const int*)d_in[29];
  }

  // ---- workspace layout ----
  char* ws = (char*)d_ws;
  size_t off = 0;
  auto take = [&](size_t bytes) -> void* {
    void* p = ws + off;
    off = (off + bytes + 255) & ~(size_t)255;
    return p;
  };
  _Float16* ebuf   = (_Float16*)take((2048L * 6400 + 1024) * 2);   // + over-read slack
  _Float16* w1f[2] = { (_Float16*)take(112L * 384 * 2), (_Float16*)take(112L * 640 * 2) };
  _Float16* w2f[2] = { (_Float16*)take(112L * 384 * 2), (_Float16*)take(112L * 640 * 2) };
  _Float16* wihp[2]= { (_Float16*)take(768L * 224 * 2), (_Float16*)take(768L * 224 * 2) };
  _Float16* whhp[2]= { (_Float16*)take(768L * 256 * 2), (_Float16*)take(768L * 256 * 2) };
  _Float16* sent   = (_Float16*)take(2048L * 224 * 2);
  float*    cbuf   = (float*)take((2048L * 48 * 112 + 256) * 4);   // conv1 & conv2 out (aliased)
  _Float16* mid    = (_Float16*)take((2048L * 48 * 128 + 1024) * 2);
  float*    part   = (float*)take((size_t)STAT_BLOCKS * 2 * 112 * 4);
  float*    scl    = (float*)take(112 * 4);
  float*    shf    = (float*)take(112 * 4);
  float*    gib[2] = { (float*)take(2048L * 768 * 4), (float*)take(2048L * 768 * 4) };
  float*    hsb[2] = { (float*)take(2048L * 256 * 4), (float*)take(2048L * 256 * 4) };

  const int Ks[2]    = { 3, 5 };
  const int Lout1[2] = { 48, 46 };   // 50-K+1
  const int Lout2[2] = { 46, 42 };   // Lout1-K+1

  // ---- weight packing (f32 -> padded f16) ----
  for (int b = 0; b < 2; ++b) {
    int tot = 112 * Ks[b] * 128;
    pack_w1<<<(tot + 255) / 256, 256, 0, stream>>>(W1[b], w1f[b], Ks[b]);
    pack_w2<<<(tot + 255) / 256, 256, 0, stream>>>(W2[b], w2f[b], Ks[b]);
    pack_wih<<<(768 * 224 + 255) / 256, 256, 0, stream>>>(WIH[b], wihp[b]);
    pack_whh<<<(768 * 256 + 255) / 256, 256, 0, stream>>>(WHH[b], whhp[b]);
  }

  // ---- embedding gather ----
  embed_gather<<<(int)((102400L * 32) / 256), 256, 0, stream>>>(X, EMB, ebuf);
  zero_h16<<<(int)((2048L * 224 + 255) / 256), 256, 0, stream>>>(sent, 2048L * 224);

  // ---- conv stacks (per kernel size, sequential; cbuf/mid reused) ----
  for (int b = 0; b < 2; ++b) {
    const int Kd = Ks[b] * 128;
    // conv1: LDS-staged sentence panel (async), WMMA GEMM
    conv_gemm_lds<<<2048, 256, 0, stream>>>(ebuf, 6400, w1f[b], Kd, cbuf, 48L * 112);
    bn_stats<<<STAT_BLOCKS, 128, 0, stream>>>(cbuf, Lout1[b], part);
    bn_finalize<<<1, 128, 0, stream>>>(part, G1[b], BE1[b], Lout1[b], scl, shf);
    bn_apply<<<(int)((2048L * 48 * 128) / 256), 256, 0, stream>>>(cbuf, scl, shf,
                                                                  Lout1[b], mid);
    // conv2: channels padded to 128, so patches are again sliding windows
    conv_gemm_lds<<<2048, 256, 0, stream>>>(mid, 48L * 128, w2f[b], Kd, cbuf, 48L * 112);
    bn_stats<<<STAT_BLOCKS, 128, 0, stream>>>(cbuf, Lout2[b], part);
    bn_finalize<<<1, 128, 0, stream>>>(part, G2[b], BE2[b], Lout2[b], scl, shf);
    bn_pool<<<(2048 * 100 + 255) / 256, 256, 0, stream>>>(cbuf, scl, shf,
                                                          Lout2[b], b * 100, sent);
  }

  // ---- GRU input projections: gi = sent . Wih^T  (2048 x 768, K=224) ----
  for (int d = 0; d < 2; ++d)
    gemm_bt<<<dim3(768), 256, 0, stream>>>(sent, 224, wihp[d], 224,
                                           gib[d], 768, 2048, 768, 224);

  // ---- persistent bidirectional scan ----
  gru_scan<<<2, 512, 0, stream>>>(gib[0], gib[1], whhp[0], whhp[1],
                                  BIH[0], BHH[0], BIH[1], BHH[1],
                                  hsb[0], hsb[1]);

  // ---- logits + sigmoid ----
  logits_kernel<<<8, 256, 0, stream>>>(hsb[0], hsb[1], CW, BIAS, (float*)d_out);
}
